// LigandGraphEncoder_17583596110570
// MI455X (gfx1250) — matrix operand order repaired and checked by reading, not compiled, not run on hardware
//
#include <hip/hip_runtime.h>
#include <math.h>

// ---------------- problem sizes ----------------
static constexpr int Hdim  = 128;
static constexpr int Bn    = 2048;
static constexpr int NAn   = 98304;
static constexpr int EAn   = 196608;
static constexpr int NCn   = 32768;
static constexpr int ECn   = 65536;
static constexpr int EXn   = 65536;
static constexpr int CBDn  = 768;
static constexpr int ATOMF = 43;
#define EPSV 1e-5f

typedef __attribute__((ext_vector_type(16))) _Float16 v16h;
typedef __attribute__((ext_vector_type(8)))  _Float16 v8h;
typedef __attribute__((ext_vector_type(8)))  float    v8f;

// ---------------- device helpers ----------------
__device__ __forceinline__ float waveSum(float v) {
  v += __shfl_xor(v, 16, 32); v += __shfl_xor(v, 8, 32);
  v += __shfl_xor(v, 4, 32);  v += __shfl_xor(v, 2, 32);
  v += __shfl_xor(v, 1, 32);
  return v;
}
__device__ __forceinline__ float waveMax(float v) {
  v = fmaxf(v, __shfl_xor(v, 16, 32)); v = fmaxf(v, __shfl_xor(v, 8, 32));
  v = fmaxf(v, __shfl_xor(v, 4, 32));  v = fmaxf(v, __shfl_xor(v, 2, 32));
  v = fmaxf(v, __shfl_xor(v, 1, 32));
  return v;
}
__device__ __forceinline__ float atomicMaxF32(float* a, float val) {
  unsigned* ua = (unsigned*)a;
  unsigned old = *ua;
  while (__uint_as_float(old) < val) {
    unsigned assumed = old;
    old = atomicCAS(ua, assumed, __float_as_uint(val));
    if (old == assumed) break;
  }
  return __uint_as_float(old);
}

// ---------------- generic fills / copies ----------------
__global__ void k_fill(float* __restrict__ p, float v, long n) {
  long i = (long)blockIdx.x * 256 + threadIdx.x;
  long stride = (long)gridDim.x * 256;
  for (; i < n; i += stride) p[i] = v;
}
__global__ void k_copy(float* __restrict__ d, const float* __restrict__ s, long n) {
  long i = (long)blockIdx.x * 256 + threadIdx.x;
  if (i < n) d[i] = s[i];
}
__global__ void k_rowbias(float* __restrict__ d, const float* __restrict__ bias, long n128) {
  long i = (long)blockIdx.x * 256 + threadIdx.x;
  if (i < n128) d[i] = bias[i & 127];
}

// ---------------- weight packing: f32 [K x N] -> CDNA5 WMMA B-fragment f16 ----------------
// B-fragment (16x16x32 f16): lane l supplies column n = l&15; half h supplies k = 8*(l>>4) + (h&7) + 16*(h>>3)
// fragment storage: dst[((chunk*ntiles + tile)*32 + lane)*16 + h]  (32B contiguous per lane)
__global__ void k_pack(const float* __restrict__ W, int K, int N, int Kp, _Float16* __restrict__ dst) {
  int frag = blockIdx.x;
  int l = threadIdx.x;
  int ntiles = N >> 4;
  int c = frag / ntiles;
  int tile = frag - c * ntiles;
  int n = (tile << 4) + (l & 15);
  _Float16* o = dst + ((size_t)frag * 32 + l) * 16;
#pragma unroll
  for (int h = 0; h < 16; ++h) {
    int k = (c << 5) + ((l >> 4) << 3) + (h & 7) + ((h >> 3) << 4);
    float v = (k < K) ? W[(size_t)k * N + n] : 0.f;
    o[h] = (_Float16)v;
  }
}

// ---------------- WMMA GEMM: C[MxN] = act(A[MxK] @ W + bias) ----------------
// block = 256 threads = 8 waves; block owns 32 rows (two 16-row M-tiles).
// Each wave owns n-tiles w, w+8, ... and computes BOTH M-tiles for each n-tile,
// so every packed-B fragment load from L2 feeds two v_wmma ops.
// A staged in LDS as f16 row-major [32 x Kp]; A-fragment K-map (CDNA5 ISA):
//   lane l (M = l&15), half h -> k = 8*(l>>4) + (h&7) + 16*(h>>3)   => two 16B LDS loads
template <int TPW>
__global__ __launch_bounds__(256) void k_gemm(
    const float* __restrict__ A, int K, int Kp,
    const _Float16* __restrict__ Wp, const float* __restrict__ bias,
    float* __restrict__ C, int N, int act) {
  extern __shared__ _Float16 ldsA[];
  const int row0 = blockIdx.x << 5;
  const int tot = Kp << 5;
  for (int i = threadIdx.x; i < tot; i += 256) {
    int r = i / Kp;
    int k = i - r * Kp;
    float v = (k < K) ? A[(size_t)(row0 + r) * K + k] : 0.f;
    ldsA[r * Kp + k] = (_Float16)v;
  }
  __syncthreads();

  const int lane = threadIdx.x & 31;
  const int w = threadIdx.x >> 5;
  const int m = lane & 15;
  const int kb = (lane >> 4) << 3;
  const int ntiles = N >> 4;

  v8f acc0[TPW] = {};
  v8f acc1[TPW] = {};
  const int chunks = Kp >> 5;
  for (int c = 0; c < chunks; ++c) {
    const v8h* pa0 = (const v8h*)(ldsA + m * Kp + (c << 5) + kb);
    const v8h* pa1 = (const v8h*)(ldsA + (m + 16) * Kp + (c << 5) + kb);
    v16h a0 = __builtin_shufflevector(pa0[0], pa0[2],
                                      0, 1, 2, 3, 4, 5, 6, 7, 8, 9, 10, 11, 12, 13, 14, 15);
    v16h a1 = __builtin_shufflevector(pa1[0], pa1[2],
                                      0, 1, 2, 3, 4, 5, 6, 7, 8, 9, 10, 11, 12, 13, 14, 15);
#pragma unroll
    for (int t = 0; t < TPW; ++t) {
      int tile = w + (t << 3);
      v16h b = *(const v16h*)(Wp + (((size_t)c * ntiles + tile) * 32 + lane) * 16);
      acc0[t] = __builtin_amdgcn_wmma_f32_16x16x32_f16(
          false, a0, false, b, (short)0, acc0[t], false, false);
      acc1[t] = __builtin_amdgcn_wmma_f32_16x16x32_f16(
          false, a1, false, b, (short)0, acc1[t], false, false);
    }
  }
#pragma unroll
  for (int t = 0; t < TPW; ++t) {
    int tile = w + (t << 3);
    int col = (tile << 4) + (lane & 15);
    float bv = bias ? bias[col] : 0.f;
#pragma unroll
    for (int r = 0; r < 8; ++r) {
      int row = row0 + r + ((lane >> 4) << 3);
      float v0 = acc0[t][r] + bv;
      float v1 = acc1[t][r] + bv;
      if (act == 1) {
        v0 = fmaxf(v0, 0.f);
        v1 = fmaxf(v1, 0.f);
      } else if (act == 2) {
        v0 = 0.5f * v0 * (1.f + erff(v0 * 0.70710678f));
        v1 = 0.5f * v1 * (1.f + erff(v1 * 0.70710678f));
      }
      C[(size_t)row * N + col] = v0;
      C[(size_t)(row + 16) * N + col] = v1;
    }
  }
}

// ---------------- column stats (C=128): sum & sumsq via per-block partials ----------------
__global__ void k_colstats(const float* __restrict__ X, int n, float* __restrict__ st) {
  int j = threadIdx.x;  // 128 threads
  int r0 = blockIdx.x * 256;
  int r1 = min(r0 + 256, n);
  float s = 0.f, sq = 0.f;
  for (int r = r0; r < r1; ++r) {
    float v = X[(size_t)r * 128 + j];
    s += v; sq += v * v;
  }
  atomicAdd(&st[j], s);
  atomicAdd(&st[128 + j], sq);
}
__global__ void k_bn_apply(const float* __restrict__ X, float* __restrict__ dst, long n128,
                           const float* __restrict__ st, const float* __restrict__ g,
                           const float* __restrict__ b, int relu, float invn) {
  long i = (long)blockIdx.x * 256 + threadIdx.x;
  if (i >= n128) return;
  int c = (int)(i & 127);
  float mean = st[c] * invn;
  float var = st[128 + c] * invn - mean * mean;
  float v = (X[i] - mean) * rsqrtf(var + EPSV) * g[c] + b[c];
  if (relu) v = fmaxf(v, 0.f);
  dst[i] = v;
}
// graphnorm: pass1 colstats (sum) -> pass2 center (in place, accumulate sq into st[256..383])
__global__ void k_gn_center(float* __restrict__ X, int n, float invn,
                            const float* __restrict__ ms, float* __restrict__ st) {
  int j = threadIdx.x;
  int r0 = blockIdx.x * 256;
  int r1 = min(r0 + 256, n);
  float mean = st[j] * invn;
  float msj = ms[j];
  float sq = 0.f;
  for (int r = r0; r < r1; ++r) {
    size_t idx = (size_t)r * 128 + j;
    float v = X[idx] - mean * msj;
    X[idx] = v;
    sq += v * v;
  }
  atomicAdd(&st[256 + j], sq);
}
__global__ void k_gn_finish(const float* __restrict__ X, const float* __restrict__ res,
                            float* __restrict__ dst, long n128, const float* __restrict__ st,
                            const float* __restrict__ w, const float* __restrict__ b, float invn) {
  long i = (long)blockIdx.x * 256 + threadIdx.x;
  if (i >= n128) return;
  int c = (int)(i & 127);
  float inv = rsqrtf(st[256 + c] * invn + EPSV);
  float v = w[c] * X[i] * inv + b[c] + res[i];
  dst[i] = v > 0.f ? v : expm1f(v);  // elu
}

// ---------------- tiny "K=1" linear: out[i,j] = act(x[i]*w[j] + b[j]) ----------------
__global__ void k_scale_vec(const float* __restrict__ x, const float* __restrict__ w,
                            const float* __restrict__ b, float* __restrict__ out,
                            long n, int C, int relu) {
  long idx = (long)blockIdx.x * 256 + threadIdx.x;
  if (idx >= n * (long)C) return;
  long i = idx / C;
  int j = (int)(idx - i * C);
  float v = x[i] * w[j] + b[j];
  if (relu) v = fmaxf(v, 0.f);
  out[idx] = v;
}

// ---------------- GATv2 (wave per edge) ----------------
__global__ void k_gat_logit(const float* __restrict__ xl, const float* __restrict__ xr,
                            const float* __restrict__ eproj, const float* __restrict__ ew,
                            const float* __restrict__ Wevec, const float* __restrict__ att,
                            const int* __restrict__ src, const int* __restrict__ dst,
                            float* __restrict__ logit, float* __restrict__ lmax, int nE) {
  int wid = (blockIdx.x * 256 + threadIdx.x) >> 5;
  int lane = threadIdx.x & 31;
  if (wid >= nE) return;
  int s = src[wid], d = dst[wid];
  float acc = 0.f;
#pragma unroll
  for (int j = 0; j < 4; ++j) {
    int c = lane + 32 * j;
    float ep = eproj ? eproj[(size_t)wid * Hdim + c] : ew[wid] * Wevec[c];
    float v = xl[(size_t)s * Hdim + c] + xr[(size_t)d * Hdim + c] + ep;
    v = v > 0.f ? v : 0.2f * v;  // leaky_relu 0.2
    acc += v * att[c];
  }
  acc = waveSum(acc);
  if (lane == 0) {
    logit[wid] = acc;
    atomicMaxF32(&lmax[d], acc);
  }
}
__global__ void k_gat_expsum(const float* __restrict__ logit, const int* __restrict__ dst,
                             const float* __restrict__ lmax, float* __restrict__ ae,
                             float* __restrict__ den, int nE) {
  int i = blockIdx.x * 256 + threadIdx.x;
  if (i >= nE) return;
  float a = expf(logit[i] - lmax[dst[i]]);
  ae[i] = a;
  atomicAdd(&den[dst[i]], a);
}
__global__ void k_gat_agg(const float* __restrict__ xl, const int* __restrict__ src,
                          const int* __restrict__ dst, const float* __restrict__ ae,
                          const float* __restrict__ den, float* __restrict__ g, int nE) {
  int wid = (blockIdx.x * 256 + threadIdx.x) >> 5;
  int lane = threadIdx.x & 31;
  if (wid >= nE) return;
  int s = src[wid], d = dst[wid];
  float coef = ae[wid] / den[d];
#pragma unroll
  for (int j = 0; j < 4; ++j) {
    int c = lane + 32 * j;
    atomicAdd(&g[(size_t)d * Hdim + c], coef * xl[(size_t)s * Hdim + c]);
  }
}
__global__ void k_cross_agg(const float* __restrict__ cl, const int* __restrict__ src,
                            const int* __restrict__ dst, float* __restrict__ agg, int nE) {
  int wid = (blockIdx.x * 256 + threadIdx.x) >> 5;
  int lane = threadIdx.x & 31;
  if (wid >= nE) return;
  int s = src[wid], d = dst[wid];
#pragma unroll
  for (int j = 0; j < 4; ++j) {
    int c = lane + 32 * j;
    atomicAdd(&agg[(size_t)d * Hdim + c], cl[(size_t)s * Hdim + c]);
  }
}

// ---------------- pooling ----------------
__global__ void k_rowdot(const float* __restrict__ X, const float* __restrict__ w,
                         const float* __restrict__ b, float* __restrict__ out, int rows) {
  int wid = (blockIdx.x * 256 + threadIdx.x) >> 5;
  int lane = threadIdx.x & 31;
  if (wid >= rows) return;
  float acc = 0.f;
#pragma unroll
  for (int j = 0; j < 4; ++j) {
    int c = lane + 32 * j;
    acc += X[(size_t)wid * Hdim + c] * w[c];
  }
  acc = waveSum(acc);
  if (lane == 0) out[wid] = acc + b[0];
}
__global__ void k_reduce_max(const float* __restrict__ x, int n, float* __restrict__ out) {
  int i = blockIdx.x * 256 + threadIdx.x;
  float v = (i < n) ? x[i] : -__builtin_inff();
  v = waveMax(v);
  if ((threadIdx.x & 31) == 0) atomicMaxF32(out, v);
}
__global__ void k_expsum(float* __restrict__ x, int n, float* __restrict__ scal) {
  int i = blockIdx.x * 256 + threadIdx.x;
  float v = 0.f;
  if (i < n) {
    v = expf(x[i] - scal[0]);
    x[i] = v;
  }
  v = waveSum(v);
  if ((threadIdx.x & 31) == 0) atomicAdd(&scal[1], v);
}
__global__ void k_pool_init(float* __restrict__ p3, long n) {  // n = B*384
  long i = (long)blockIdx.x * 256 + threadIdx.x;
  if (i >= n) return;
  int c = (int)(i % 384);
  p3[i] = (c >= 128 && c < 256) ? -__builtin_inff() : 0.f;
}
__global__ void k_pool_accum(const float* __restrict__ X, const int* __restrict__ batch,
                             const float* __restrict__ score, const float* __restrict__ scal,
                             float* __restrict__ p3, float* __restrict__ cnt, int rows) {
  int wid = (blockIdx.x * 256 + threadIdx.x) >> 5;
  int lane = threadIdx.x & 31;
  if (wid >= rows) return;
  int b = batch[wid];
  float wgt = score[wid] / scal[1];
#pragma unroll
  for (int j = 0; j < 4; ++j) {
    int c = lane + 32 * j;
    float xv = X[(size_t)wid * Hdim + c];
    atomicAdd(&p3[(size_t)b * 384 + c], xv * wgt);
    atomicMaxF32(&p3[(size_t)b * 384 + 128 + c], xv);
    atomicAdd(&p3[(size_t)b * 384 + 256 + c], xv);
  }
  if (lane == 0) atomicAdd(&cnt[b], 1.f);
}
__global__ void k_pool_finish(float* __restrict__ p3, const float* __restrict__ cnt, long nBH) {
  long i = (long)blockIdx.x * 256 + threadIdx.x;
  if (i >= nBH) return;
  int b = (int)(i >> 7);
  int c = (int)(i & 127);
  float ct = fmaxf(cnt[b], 1.f);
  p3[(size_t)b * 384 + 256 + c] /= ct;
  float mv = p3[(size_t)b * 384 + 128 + c];
  if (!__builtin_isfinite(mv)) p3[(size_t)b * 384 + 128 + c] = 0.f;
}

// ---------------- LayerNorm (wave per row, H=128); optional residual ----------------
__global__ void k_ln(const float* __restrict__ X, const float* __restrict__ R,
                     const float* __restrict__ g, const float* __restrict__ b,
                     float* __restrict__ out, int rows) {
  int wid = (blockIdx.x * 256 + threadIdx.x) >> 5;
  int lane = threadIdx.x & 31;
  if (wid >= rows) return;
  float v[4];
  float s = 0.f;
#pragma unroll
  for (int j = 0; j < 4; ++j) {
    int c = lane + 32 * j;
    float x = X[(size_t)wid * Hdim + c] + (R ? R[(size_t)wid * Hdim + c] : 0.f);
    v[j] = x;
    s += x;
  }
  s = waveSum(s);
  float mean = s * (1.f / 128.f);
  float sq = 0.f;
#pragma unroll
  for (int j = 0; j < 4; ++j) {
    float d = v[j] - mean;
    sq += d * d;
  }
  sq = waveSum(sq);
  float inv = rsqrtf(sq * (1.f / 128.f) + EPSV);
#pragma unroll
  for (int j = 0; j < 4; ++j) {
    int c = lane + 32 * j;
    out[(size_t)wid * Hdim + c] = (v[j] - mean) * inv * g[c] + b[c];
  }
}

// ---------------- fusion MHA over S=3 tokens, 4 heads of 32 ----------------
__global__ void k_mha3(const float* __restrict__ Q, const float* __restrict__ Km,
                       const float* __restrict__ V, float* __restrict__ O) {
  int t = blockIdx.x * 256 + threadIdx.x;
  if (t >= Bn * 4) return;
  int b = t >> 2;
  int h = t & 3;
  size_t base = (size_t)b * Hdim + h * 32;
  const size_t BH = (size_t)Bn * Hdim;
  float sc[3][3];
#pragma unroll
  for (int s = 0; s < 3; ++s)
#pragma unroll
    for (int u = 0; u < 3; ++u) {
      float d = 0.f;
      for (int k = 0; k < 32; ++k)
        d += Q[s * BH + base + k] * Km[u * BH + base + k];
      sc[s][u] = d * 0.17677669529f;  // 1/sqrt(32)
    }
  float a[3][3];
#pragma unroll
  for (int s = 0; s < 3; ++s) {
    float mx = fmaxf(sc[s][0], fmaxf(sc[s][1], sc[s][2]));
    float e0 = expf(sc[s][0] - mx), e1 = expf(sc[s][1] - mx), e2 = expf(sc[s][2] - mx);
    float inv = 1.f / (e0 + e1 + e2);
    a[s][0] = e0 * inv; a[s][1] = e1 * inv; a[s][2] = e2 * inv;
  }
  for (int k = 0; k < 32; ++k) {
    float v0 = V[0 * BH + base + k], v1 = V[1 * BH + base + k], v2 = V[2 * BH + base + k];
#pragma unroll
    for (int s = 0; s < 3; ++s)
      O[s * BH + base + k] = a[s][0] * v0 + a[s][1] * v1 + a[s][2] * v2;
  }
}
__global__ void k_concat3(const float* __restrict__ a, const float* __restrict__ b,
                          const float* __restrict__ c, float* __restrict__ out) {
  long i = (long)blockIdx.x * 256 + threadIdx.x;
  const long BH = (long)Bn * Hdim;
  if (i >= 3 * BH) return;
  int s = (int)(i / BH);
  long r = i - (long)s * BH;
  const float* p = (s == 0) ? a : (s == 1) ? b : c;
  out[i] = p[r];
}
// fused = mha_out + mean_s(attn_in); flat[b, s*H+c] = fused[s,b,c]
__global__ void k_fused_flat(const float* __restrict__ mha, const float* __restrict__ ain,
                             float* __restrict__ flat) {
  long i = (long)blockIdx.x * 256 + threadIdx.x;
  const long BH = (long)Bn * Hdim;
  if (i >= 3 * BH) return;
  int s = (int)(i / BH);
  long r = i - (long)s * BH;
  int b = (int)(r >> 7);
  int c = (int)(r & 127);
  float mean = (ain[r] + ain[BH + r] + ain[2 * BH + r]) * (1.f / 3.f);
  flat[(size_t)b * 384 + s * 128 + c] = mha[i] + mean;
}

// ---------------- GEMM dispatcher (32 rows per block) ----------------
static void launch_gemm(hipStream_t st, const float* A, int M, int K,
                        const _Float16* Wp, const float* bias, float* C, int N, int act) {
  int Kp = (K + 31) & ~31;
  size_t lds = (size_t)32 * Kp * sizeof(_Float16);
  dim3 g(M / 32), b(256);
  int nt = N >> 4;
  if (nt == 8)       k_gemm<1><<<g, b, lds, st>>>(A, K, Kp, Wp, bias, C, N, act);
  else if (nt == 16) k_gemm<2><<<g, b, lds, st>>>(A, K, Kp, Wp, bias, C, N, act);
  else               k_gemm<4><<<g, b, lds, st>>>(A, K, Kp, Wp, bias, C, N, act);
}

// ---------------- param index map (params pytree flattened with sorted keys) ----------------
enum {
  P_atom_bn_b = 0, P_atom_bn_g, P_atom_fc_b, P_atom_fc_w,
  P_cb1_b, P_cb1_w, P_cb2_b, P_cb2_w, P_cb_bn_b, P_cb_bn_g, P_cb_proj_b, P_cb_proj_w,
  P_cl_bn_b, P_cl_bn_g, P_cl_pp_b, P_cl_pp_w,
  P_clique_fc_b, P_clique_fc_w, P_clique_res_b, P_clique_res_w,
  P_cn_b, P_cn_g,
  P_cp1_b, P_cp1_w, P_cp2_b, P_cp2_w,
  P_edge1_b, P_edge1_w, P_edge2_b, P_edge2_w,
  P_f_k_b, P_f_k_w, P_f_o_b, P_f_o_w, P_f_q_b, P_f_q_w, P_f_v_b, P_f_v_w,
  P_fbn_b, P_fbn_g, P_fw_b, P_fw_w,
  P_gc_We, P_gc_att, P_gc_bias, P_gc_l_b, P_gc_l_w, P_gc_r_b, P_gc_r_w,
  P_gcn_b, P_gcn_ms, P_gcn_w,
  P_gin1_b, P_gin1_w, P_gin2_b, P_gin2_w,
  P_gm_We, P_gm_att, P_gm_bias, P_gm_l_b, P_gm_l_w, P_gm_r_b, P_gm_r_w,
  P_gmn_b, P_gmn_ms, P_gmn_w,
  P_mol_pp_b, P_mol_pp_w, P_mol_res_b, P_mol_res_w,
  P_mp1_b, P_mp1_w, P_mp2_b, P_mp2_w,
  P_raw_b, P_raw_w,
  P_t_ff1_b, P_t_ff1_w, P_t_ff2_b, P_t_ff2_w,
  P_t_k_b, P_t_k_w, P_t_ln1_b, P_t_ln1_g, P_t_ln2_b, P_t_ln2_g,
  P_t_o_b, P_t_o_w, P_t_q_b, P_t_q_w, P_t_v_b, P_t_v_w,
  P_COUNT  // 92
};

extern "C" void kernel_launch(void* const* d_in, const int* in_sizes, int n_in,
                              void* d_out, int out_size, void* d_ws, size_t ws_size,
                              hipStream_t stream) {
  (void)in_sizes; (void)n_in; (void)out_size; (void)ws_size;
#define PP(name) ((const float*)d_in[5 + P_##name])
  const float* mol_x   = (const float*)d_in[0];
  const float* mol_ea  = (const float*)d_in[1];
  const float* cli_x   = (const float*)d_in[2];
  const float* cli_ew  = (const float*)d_in[3];
  const float* chem    = (const float*)d_in[4];
  const int* mol_ei    = (const int*)d_in[5 + P_COUNT + 0];
  const int* mol_bat   = (const int*)d_in[5 + P_COUNT + 1];
  const int* cross_ei  = (const int*)d_in[5 + P_COUNT + 2];
  const int* cli_ei    = (const int*)d_in[5 + P_COUNT + 3];
  const int* cli_bat   = (const int*)d_in[5 + P_COUNT + 4];

  // ---- workspace allocator ----
  char* wsb = (char*)d_ws;
  size_t off = 0;
  auto allocF = [&](size_t n) -> float* {
    float* p = (float*)(wsb + off);
    off += ((n * sizeof(float)) + 255) & ~(size_t)255;
    return p;
  };
  auto allocH = [&](size_t n) -> _Float16* {
    _Float16* p = (_Float16*)(wsb + off);
    off += ((n * sizeof(_Float16)) + 255) & ~(size_t)255;
    return p;
  };
  const size_t BH = (size_t)Bn * Hdim;

  float* A_atom = allocF((size_t)NAn * Hdim);
  float* A_res  = allocF((size_t)NAn * Hdim);
  float* A_xl   = allocF((size_t)NAn * Hdim);
  float* A_xr   = allocF((size_t)NAn * Hdim);
  float* A_g    = allocF((size_t)NAn * Hdim);
  float* A_feat = allocF((size_t)NAn * Hdim);
  float* E_h    = allocF((size_t)EAn * 64);
  float* E_enc  = allocF((size_t)EAn * Hdim);
  float* E_proj = allocF((size_t)EAn * Hdim);
  float* e_log  = allocF(EAn);
  float* e_a    = allocF(EAn);
  float* n_max  = allocF(NAn);
  float* n_den  = allocF(NAn);
  float* n_sc   = allocF(NAn);
  float* C_cl   = allocF((size_t)NCn * Hdim);
  float* C_agg  = allocF((size_t)NCn * Hdim);
  float* C_h    = allocF((size_t)NCn * 256);
  float* C_res  = allocF((size_t)NCn * Hdim);
  float* C_xl   = allocF((size_t)NCn * Hdim);
  float* C_xr   = allocF((size_t)NCn * Hdim);
  float* C_g    = allocF((size_t)NCn * Hdim);
  float* C_feat = allocF((size_t)NCn * Hdim);
  float* c_log  = allocF(ECn);
  float* c_a    = allocF(ECn);
  float* cn_max = allocF(NCn);
  float* cn_den = allocF(NCn);
  float* cn_sc  = allocF(NCn);
  float* P_m3   = allocF((size_t)Bn * 384);
  float* P_m    = allocF(BH);
  float* P_c3   = allocF((size_t)Bn * 384);
  float* P_c    = allocF(BH);
  float* P_cnt  = allocF(Bn);
  float* CB_big = allocF((size_t)Bn * 512);
  float* CB_x   = allocF(BH);
  float* CB_v   = allocF(BH);
  float* CB_t1  = allocF(BH);
  float* CB_y   = allocF(BH);
  float* CB_z   = allocF(BH);
  float* CB_t2  = allocF(BH);
  float* CB_f   = allocF(BH);
  float* F_in   = allocF(3 * BH);
  float* F_q    = allocF(3 * BH);
  float* F_k    = allocF(3 * BH);
  float* F_v    = allocF(3 * BH);
  float* F_att  = allocF(3 * BH);
  float* F_mha  = allocF(3 * BH);
  float* F_flat = allocF((size_t)Bn * 384);
  float* stats  = allocF(512);
  float* scal   = allocF(64);
  _Float16* packBase = allocH(2200000);
  size_t packOff = 0;

  auto packW = [&](const float* W, int K, int N) -> const _Float16* {
    int Kp = (K + 31) & ~31;
    _Float16* dst = packBase + packOff;
    packOff += (size_t)Kp * N;
    int frags = (Kp >> 5) * (N >> 4);
    k_pack<<<frags, 32, 0, stream>>>(W, K, N, Kp, dst);
    return dst;
  };
  auto cd = [](long n) -> unsigned { return (unsigned)((n + 255) / 256); };

  // ---- pack all GEMM weights to f16 B-fragment layout ----
  const _Float16* Wp_atom_fc = packW(PP(atom_fc_w), ATOMF, 128);
  const _Float16* Wp_edge2   = packW(PP(edge2_w), 64, 128);
  const _Float16* Wp_molres  = packW(PP(mol_res_w), 128, 128);
  const _Float16* Wp_gml     = packW(PP(gm_l_w), 128, 128);
  const _Float16* Wp_gmr     = packW(PP(gm_r_w), 128, 128);
  const _Float16* Wp_gmWe    = packW(PP(gm_We), 128, 128);
  const _Float16* Wp_mp1     = packW(PP(mp1_w), 128, 128);
  const _Float16* Wp_molpp   = packW(PP(mol_pp_w), 384, 128);
  const _Float16* Wp_gin1    = packW(PP(gin1_w), 128, 256);
  const _Float16* Wp_gin2    = packW(PP(gin2_w), 256, 128);
  const _Float16* Wp_clres   = packW(PP(clique_res_w), 128, 128);
  const _Float16* Wp_gcl     = packW(PP(gc_l_w), 128, 128);
  const _Float16* Wp_gcr     = packW(PP(gc_r_w), 128, 128);
  const _Float16* Wp_cp1     = packW(PP(cp1_w), 128, 128);
  const _Float16* Wp_clpp    = packW(PP(cl_pp_w), 384, 128);
  const _Float16* Wp_cb1     = packW(PP(cb1_w), CBDn, 256);
  const _Float16* Wp_cb2     = packW(PP(cb2_w), 256, 128);
  const _Float16* Wp_tv      = packW(PP(t_v_w), 128, 128);
  const _Float16* Wp_to      = packW(PP(t_o_w), 128, 128);
  const _Float16* Wp_tff1    = packW(PP(t_ff1_w), 128, 512);
  const _Float16* Wp_tff2    = packW(PP(t_ff2_w), 512, 128);
  const _Float16* Wp_cbproj  = packW(PP(cb_proj_w), 128, 128);
  const _Float16* Wp_fq      = packW(PP(f_q_w), 128, 128);
  const _Float16* Wp_fk      = packW(PP(f_k_w), 128, 128);
  const _Float16* Wp_fv      = packW(PP(f_v_w), 128, 128);
  const _Float16* Wp_fo      = packW(PP(f_o_w), 128, 128);
  const _Float16* Wp_fw      = packW(PP(fw_w), 384, 128);
  const _Float16* Wp_raw     = packW(PP(raw_w), 128, 128);

  // ========== mol branch ==========
  launch_gemm(stream, mol_x, NAn, ATOMF, Wp_atom_fc, PP(atom_fc_b), A_atom, 128, 0);
  k_fill<<<1, 256, 0, stream>>>(stats, 0.f, 512);
  k_colstats<<<NAn / 256, 128, 0, stream>>>(A_atom, NAn, stats);
  k_bn_apply<<<cd((long)NAn * 128), 256, 0, stream>>>(A_atom, A_atom, (long)NAn * 128, stats,
                                                      PP(atom_bn_g), PP(atom_bn_b), 1, 1.f / NAn);
  k_scale_vec<<<cd((long)EAn * 64), 256, 0, stream>>>(mol_ea, PP(edge1_w), PP(edge1_b), E_h, EAn, 64, 1);
  launch_gemm(stream, E_h, EAn, 64, Wp_edge2, PP(edge2_b), E_enc, 128, 0);
  launch_gemm(stream, A_atom, NAn, 128, Wp_molres, PP(mol_res_b), A_res, 128, 0);
  launch_gemm(stream, A_atom, NAn, 128, Wp_gml, PP(gm_l_b), A_xl, 128, 0);
  launch_gemm(stream, A_atom, NAn, 128, Wp_gmr, PP(gm_r_b), A_xr, 128, 0);
  launch_gemm(stream, E_enc, EAn, 128, Wp_gmWe, nullptr, E_proj, 128, 0);
  // GATv2 (mol)
  k_fill<<<cd(NAn), 256, 0, stream>>>(n_max, -__builtin_inff(), NAn);
  k_gat_logit<<<EAn / 8, 256, 0, stream>>>(A_xl, A_xr, E_proj, nullptr, nullptr, PP(gm_att),
                                           mol_ei, mol_ei + EAn, e_log, n_max, EAn);
  k_fill<<<cd(NAn), 256, 0, stream>>>(n_den, 0.f, NAn);
  k_gat_expsum<<<cd(EAn), 256, 0, stream>>>(e_log, mol_ei + EAn, n_max, e_a, n_den, EAn);
  k_rowbias<<<cd((long)NAn * 128), 256, 0, stream>>>(A_g, PP(gm_bias), (long)NAn * 128);
  k_gat_agg<<<EAn / 8, 256, 0, stream>>>(A_xl, mol_ei, mol_ei + EAn, e_a, n_den, A_g, EAn);
  // graphnorm + elu(g + res)
  k_fill<<<1, 256, 0, stream>>>(stats, 0.f, 512);
  k_colstats<<<NAn / 256, 128, 0, stream>>>(A_g, NAn, stats);
  k_gn_center<<<NAn / 256, 128, 0, stream>>>(A_g, NAn, 1.f / NAn, PP(gmn_ms), stats);
  k_gn_finish<<<cd((long)NAn * 128), 256, 0, stream>>>(A_g, A_res, A_feat, (long)NAn * 128, stats,
                                                       PP(gmn_w), PP(gmn_b), 1.f / NAn);
  // pool (mol) — softmax over the full score vector, as in the reference
  launch_gemm(stream, A_feat, NAn, 128, Wp_mp1, PP(mp1_b), A_xl, 128, 1);
  k_rowdot<<<NAn / 8, 256, 0, stream>>>(A_xl, PP(mp2_w), PP(mp2_b), n_sc, NAn);
  k_fill<<<1, 256, 0, stream>>>(scal, -__builtin_inff(), 1);
  k_reduce_max<<<cd(NAn), 256, 0, stream>>>(n_sc, NAn, scal);
  k_fill<<<1, 256, 0, stream>>>(scal + 1, 0.f, 1);
  k_expsum<<<cd(NAn), 256, 0, stream>>>(n_sc, NAn, scal);
  k_pool_init<<<cd((long)Bn * 384), 256, 0, stream>>>(P_m3, (long)Bn * 384);
  k_fill<<<cd(Bn), 256, 0, stream>>>(P_cnt, 0.f, Bn);
  k_pool_accum<<<NAn / 8, 256, 0, stream>>>(A_feat, mol_bat, n_sc, scal, P_m3, P_cnt, NAn);
  k_pool_finish<<<cd((long)Bn * 128), 256, 0, stream>>>(P_m3, P_cnt, (long)Bn * 128);
  launch_gemm(stream, P_m3, Bn, 384, Wp_molpp, PP(mol_pp_b), P_m, 128, 0);

  // ========== clique branch ==========
  k_scale_vec<<<cd((long)NCn * 128), 256, 0, stream>>>(cli_x, PP(clique_fc_w), PP(clique_fc_b),
                                                       C_cl, NCn, 128, 0);
  k_fill<<<1, 256, 0, stream>>>(stats, 0.f, 512);
  k_colstats<<<NCn / 256, 128, 0, stream>>>(C_cl, NCn, stats);
  k_bn_apply<<<cd((long)NCn * 128), 256, 0, stream>>>(C_cl, C_cl, (long)NCn * 128, stats,
                                                      PP(cl_bn_g), PP(cl_bn_b), 1, 1.f / NCn);
  k_copy<<<cd((long)NCn * 128), 256, 0, stream>>>(C_agg, C_cl, (long)NCn * 128);
  k_cross_agg<<<EXn / 8, 256, 0, stream>>>(C_cl, cross_ei, cross_ei + EXn, C_agg, EXn);
  launch_gemm(stream, C_agg, NCn, 128, Wp_gin1, PP(gin1_b), C_h, 256, 1);
  launch_gemm(stream, C_h, NCn, 256, Wp_gin2, PP(gin2_b), C_cl, 128, 0);
  launch_gemm(stream, C_cl, NCn, 128, Wp_clres, PP(clique_res_b), C_res, 128, 0);
  launch_gemm(stream, C_cl, NCn, 128, Wp_gcl, PP(gc_l_b), C_xl, 128, 0);
  launch_gemm(stream, C_cl, NCn, 128, Wp_gcr, PP(gc_r_b), C_xr, 128, 0);
  // GATv2 (clique): e_proj_e = weight_e * gc_We_row
  k_fill<<<cd(NCn), 256, 0, stream>>>(cn_max, -__builtin_inff(), NCn);
  k_gat_logit<<<ECn / 8, 256, 0, stream>>>(C_xl, C_xr, nullptr, cli_ew, PP(gc_We), PP(gc_att),
                                           cli_ei, cli_ei + ECn, c_log, cn_max, ECn);
  k_fill<<<cd(NCn), 256, 0, stream>>>(cn_den, 0.f, NCn);
  k_gat_expsum<<<cd(ECn), 256, 0, stream>>>(c_log, cli_ei + ECn, cn_max, c_a, cn_den, ECn);
  k_rowbias<<<cd((long)NCn * 128), 256, 0, stream>>>(C_g, PP(gc_bias), (long)NCn * 128);
  k_gat_agg<<<ECn / 8, 256, 0, stream>>>(C_xl, cli_ei, cli_ei + ECn, c_a, cn_den, C_g, ECn);
  k_fill<<<1, 256, 0, stream>>>(stats, 0.f, 512);
  k_colstats<<<NCn / 256, 128, 0, stream>>>(C_g, NCn, stats);
  k_gn_center<<<NCn / 256, 128, 0, stream>>>(C_g, NCn, 1.f / NCn, PP(gcn_ms), stats);
  k_gn_finish<<<cd((long)NCn * 128), 256, 0, stream>>>(C_g, C_res, C_feat, (long)NCn * 128, stats,
                                                       PP(gcn_w), PP(gcn_b), 1.f / NCn);
  // pool (clique)
  launch_gemm(stream, C_feat, NCn, 128, Wp_cp1, PP(cp1_b), C_xl, 128, 1);
  k_rowdot<<<NCn / 8, 256, 0, stream>>>(C_xl, PP(cp2_w), PP(cp2_b), cn_sc, NCn);
  k_fill<<<1, 256, 0, stream>>>(scal, -__builtin_inff(), 1);
  k_reduce_max<<<cd(NCn), 256, 0, stream>>>(cn_sc, NCn, scal);
  k_fill<<<1, 256, 0, stream>>>(scal + 1, 0.f, 1);
  k_expsum<<<cd(NCn), 256, 0, stream>>>(cn_sc, NCn, scal);
  k_pool_init<<<cd((long)Bn * 384), 256, 0, stream>>>(P_c3, (long)Bn * 384);
  k_fill<<<cd(Bn), 256, 0, stream>>>(P_cnt, 0.f, Bn);
  k_pool_accum<<<NCn / 8, 256, 0, stream>>>(C_feat, cli_bat, cn_sc, scal, P_c3, P_cnt, NCn);
  k_pool_finish<<<cd((long)Bn * 128), 256, 0, stream>>>(P_c3, P_cnt, (long)Bn * 128);
  launch_gemm(stream, P_c3, Bn, 384, Wp_clpp, PP(cl_pp_b), P_c, 128, 0);

  // ========== chemBERTa branch ==========
  launch_gemm(stream, chem, Bn, CBDn, Wp_cb1, PP(cb1_b), CB_big, 256, 1);
  launch_gemm(stream, CB_big, Bn, 256, Wp_cb2, PP(cb2_b), CB_x, 128, 0);
  k_fill<<<1, 256, 0, stream>>>(stats, 0.f, 512);
  k_colstats<<<Bn / 256, 128, 0, stream>>>(CB_x, Bn, stats);
  k_bn_apply<<<cd((long)Bn * 128), 256, 0, stream>>>(CB_x, CB_x, (long)Bn * 128, stats,
                                                     PP(cb_bn_g), PP(cb_bn_b), 1, 1.f / Bn);
  // transformer, S=1: softmax over one token == 1, so MHA(x) = (x@Wv+bv)@Wo+bo
  launch_gemm(stream, CB_x, Bn, 128, Wp_tv, PP(t_v_b), CB_v, 128, 0);
  launch_gemm(stream, CB_v, Bn, 128, Wp_to, PP(t_o_b), CB_t1, 128, 0);
  k_ln<<<Bn / 8, 256, 0, stream>>>(CB_x, CB_t1, PP(t_ln1_g), PP(t_ln1_b), CB_y, Bn);
  launch_gemm(stream, CB_y, Bn, 128, Wp_tff1, PP(t_ff1_b), CB_big, 512, 2);  // exact gelu
  launch_gemm(stream, CB_big, Bn, 512, Wp_tff2, PP(t_ff2_b), CB_t1, 128, 0);
  k_ln<<<Bn / 8, 256, 0, stream>>>(CB_y, CB_t1, PP(t_ln2_g), PP(t_ln2_b), CB_z, Bn);
  k_ln<<<Bn / 8, 256, 0, stream>>>(CB_z, nullptr, PP(cn_g), PP(cn_b), CB_t2, Bn);
  launch_gemm(stream, CB_t2, Bn, 128, Wp_cbproj, PP(cb_proj_b), CB_f, 128, 0);

  // ========== fusion ==========
  k_concat3<<<cd(3 * (long)BH), 256, 0, stream>>>(P_m, P_c, CB_f, F_in);
  launch_gemm(stream, F_in, 3 * Bn, 128, Wp_fq, PP(f_q_b), F_q, 128, 0);
  launch_gemm(stream, F_in, 3 * Bn, 128, Wp_fk, PP(f_k_b), F_k, 128, 0);
  launch_gemm(stream, F_in, 3 * Bn, 128, Wp_fv, PP(f_v_b), F_v, 128, 0);
  k_mha3<<<cd((long)Bn * 4), 256, 0, stream>>>(F_q, F_k, F_v, F_att);
  launch_gemm(stream, F_att, 3 * Bn, 128, Wp_fo, PP(f_o_b), F_mha, 128, 0);
  k_fused_flat<<<cd(3 * (long)BH), 256, 0, stream>>>(F_mha, F_in, F_flat);
  launch_gemm(stream, F_flat, Bn, 384, Wp_fw, PP(fw_b), CB_t1, 128, 0);
  k_fill<<<1, 256, 0, stream>>>(stats, 0.f, 512);
  k_colstats<<<Bn / 256, 128, 0, stream>>>(CB_t1, Bn, stats);
  k_bn_apply<<<cd((long)Bn * 128), 256, 0, stream>>>(CB_t1, (float*)d_out, (long)Bn * 128, stats,
                                                     PP(fbn_g), PP(fbn_b), 0, 1.f / Bn);
  // raw = mol_feat @ raw_w + raw_b  -> second output
  launch_gemm(stream, A_feat, NAn, 128, Wp_raw, PP(raw_b), (float*)d_out + BH, 128, 0);
#undef PP
}
